// SelfAttentionLayer_23141283790864
// MI455X (gfx1250) — compile-verified, matrix operand order checked by reading, not run on hardware
//
#include <hip/hip_runtime.h>
#include <hip/hip_bf16.h>

// ---------------------------------------------------------------------------
// Self-attention (B=4, N=2048, D=1024) on gfx1250 via v_wmma_f32_16x16x32_bf16
// Pipeline: f32->bf16 staging -> QKV proj -> Q.K^T -> softmax -> P.V
// ---------------------------------------------------------------------------

#define BB    4
#define NTOK  2048
#define DIN   1024
#define DOUT  1024
#define MTOT  (BB * NTOK)          // 8192 flattened token rows

typedef __attribute__((ext_vector_type(16))) __bf16 v16bf;
typedef __attribute__((ext_vector_type(8)))  __bf16 v8bf;
typedef __attribute__((ext_vector_type(4)))  __bf16 v4bf;
typedef __attribute__((ext_vector_type(8)))  float  v8f;
typedef __attribute__((ext_vector_type(4)))  float  v4f;

static __device__ __forceinline__ __bf16 f2bf(float f) { return (__bf16)f; }

// WMMA A/B operand fragment from a bf16 row (16x32 tile operand):
// elements 0..7 <- base[0..7], elements 8..15 <- base[16..23]
// (ISA 7.12.2: per-lane k = (e>>3)*16 + khalf + (e&7); khalf folded into base)
static __device__ __forceinline__ v16bf load_frag_bf16(const __bf16* __restrict__ base) {
    v8bf lo = *(const v8bf*)(base);
    v8bf hi = *(const v8bf*)(base + 16);
    return __builtin_shufflevector(lo, hi, 0, 1, 2, 3, 4, 5, 6, 7,
                                           8, 9, 10, 11, 12, 13, 14, 15);
}

static __device__ __forceinline__ v8f wmma_bf16(v16bf a, v16bf b, v8f c) {
    // (neg_a, A, neg_b, B, c_mod, C, reuse_a, reuse_b)
    return __builtin_amdgcn_wmma_f32_16x16x32_bf16(false, a, false, b, (short)0, c, false, false);
}

// ---------------------------------------------------------------------------
// Kernel 0: flat f32 -> bf16 conversion, 8 elements/thread (16B stores).
// Native v_cvt path via vector converts; runs once per tensor.
// ---------------------------------------------------------------------------
__global__ __launch_bounds__(256) void cvt_f32_to_bf16(
    const float* __restrict__ in, __bf16* __restrict__ outp, int n8)
{
    const int i = blockIdx.x * 256 + threadIdx.x;
    if (i >= n8) return;
    const v4f* __restrict__ p = (const v4f*)in + 2 * (size_t)i;
    v4bf lo = __builtin_convertvector(p[0], v4bf);
    v4bf hi = __builtin_convertvector(p[1], v4bf);
    *((v8bf*)outp + i) = __builtin_shufflevector(lo, hi, 0, 1, 2, 3, 4, 5, 6, 7);
}

// ---------------------------------------------------------------------------
// Kernel 1: QKV projection.  Y = Xb @ Wb^T (all bf16 in, bf16 out).
// Wb is [out,in] row-major -> B-operand rows are contiguous in k.
// blockIdx.y: 0=Q, 1=K, 2=V (V stored transposed Vt[b][d][tok]).
// Per wave: 32x64 output tile (2 A frags x 4 B frags = 8 wmma / k-step).
// ---------------------------------------------------------------------------
__global__ __launch_bounds__(256) void attn_qkv_proj(
    const __bf16* __restrict__ xb,
    const __bf16* __restrict__ Wqb, const __bf16* __restrict__ Wkb,
    const __bf16* __restrict__ Wvb,
    __bf16* __restrict__ Qb, __bf16* __restrict__ Kb, __bf16* __restrict__ Vt)
{
    const int lane  = threadIdx.x & 31;
    const int wave  = threadIdx.x >> 5;
    const int which = blockIdx.y;
    const __bf16* __restrict__ W = (which == 0) ? Wqb : ((which == 1) ? Wkb : Wvb);

    const int TN = DOUT / 64;                   // 16 column tiles
    const int t  = blockIdx.x * 8 + wave;       // 0..4095 wave tiles
    const int tm = t / TN;                      // 0..255 (32-row tiles)
    const int tn = t % TN;

    const int m15   = lane & 15;
    const int khalf = (lane >> 4) << 3;         // 0 or 8

    const __bf16* __restrict__ arow0 = xb + (size_t)(tm * 32      + m15) * DIN + khalf;
    const __bf16* __restrict__ arow1 = xb + (size_t)(tm * 32 + 16 + m15) * DIN + khalf;

    v8f acc[2][4] = {};
    for (int k = 0; k < DIN; k += 32) {
        __builtin_prefetch(arow0 + k + 128, 0, 0);
        v16bf a0 = load_frag_bf16(arow0 + k);
        v16bf a1 = load_frag_bf16(arow1 + k);
#pragma unroll
        for (int j = 0; j < 4; ++j) {
            const __bf16* __restrict__ brow =
                W + (size_t)(tn * 64 + j * 16 + m15) * DIN + khalf + k;
            v16bf b = load_frag_bf16(brow);
            acc[0][j] = wmma_bf16(a0, b, acc[0][j]);
            acc[1][j] = wmma_bf16(a1, b, acc[1][j]);
        }
    }

    if (which == 2) {
        // V: transposed per-batch store Vt[b][d][tok] (32 rows never cross batch)
        const int batch = (tm * 32) / NTOK;
        const int tokb  = (tm * 32) % NTOK + khalf;
#pragma unroll
        for (int i = 0; i < 2; ++i) {
#pragma unroll
            for (int j = 0; j < 4; ++j) {
                const int d = tn * 64 + j * 16 + m15;
                __bf16* __restrict__ p =
                    Vt + ((size_t)batch * DOUT + d) * NTOK + tokb + i * 16;
#pragma unroll
                for (int r = 0; r < 8; ++r) p[r] = f2bf(acc[i][j][r]);
            }
        }
    } else {
        __bf16* __restrict__ Y = (which == 0) ? Qb : Kb;
#pragma unroll
        for (int i = 0; i < 2; ++i) {
#pragma unroll
            for (int j = 0; j < 4; ++j) {
                const int d = tn * 64 + j * 16 + m15;
#pragma unroll
                for (int r = 0; r < 8; ++r) {
                    const int m = tm * 32 + i * 16 + khalf + r;
                    Y[(size_t)m * DOUT + d] = f2bf(acc[i][j][r]);
                }
            }
        }
    }
}

// ---------------------------------------------------------------------------
// Kernel 2: scores S = (Q @ K^T) / sqrt(D), f32 out.  32x64 tile per wave.
// ---------------------------------------------------------------------------
__global__ __launch_bounds__(256) void attn_scores(
    const __bf16* __restrict__ Qb, const __bf16* __restrict__ Kb,
    float* __restrict__ S)
{
    const int lane = threadIdx.x & 31;
    const int wave = threadIdx.x >> 5;

    const int TPB   = (NTOK / 32) * (NTOK / 64);    // 2048 tiles per batch
    const int t     = blockIdx.x * 8 + wave;
    const int batch = t / TPB;
    const int rem   = t % TPB;
    const int tm    = rem >> 5;                     // 0..63 (32-row tiles)
    const int tn    = rem & 31;                     // 0..31 (64-col tiles)

    const int m15   = lane & 15;
    const int khalf = (lane >> 4) << 3;

    const __bf16* __restrict__ arow0 =
        Qb + (size_t)(batch * NTOK + tm * 32      + m15) * DIN + khalf;
    const __bf16* __restrict__ arow1 =
        Qb + (size_t)(batch * NTOK + tm * 32 + 16 + m15) * DIN + khalf;

    v8f acc[2][4] = {};
    for (int k = 0; k < DIN; k += 32) {
        __builtin_prefetch(arow0 + k + 128, 0, 0);
        v16bf a0 = load_frag_bf16(arow0 + k);
        v16bf a1 = load_frag_bf16(arow1 + k);
#pragma unroll
        for (int j = 0; j < 4; ++j) {
            const __bf16* __restrict__ brow =
                Kb + (size_t)(batch * NTOK + tn * 64 + j * 16 + m15) * DIN + khalf + k;
            v16bf b = load_frag_bf16(brow);
            acc[0][j] = wmma_bf16(a0, b, acc[0][j]);
            acc[1][j] = wmma_bf16(a1, b, acc[1][j]);
        }
    }

    const float scale = 0.03125f; // 1/sqrt(1024)
#pragma unroll
    for (int i = 0; i < 2; ++i) {
#pragma unroll
        for (int j = 0; j < 4; ++j) {
            const int jc = tn * 64 + j * 16 + m15;
#pragma unroll
            for (int r = 0; r < 8; ++r) {
                const int ir = tm * 32 + i * 16 + khalf + r;
                S[((size_t)batch * NTOK + ir) * NTOK + jc] = acc[i][j][r] * scale;
            }
        }
    }
}

// ---------------------------------------------------------------------------
// Kernel 3: row softmax over 2048 f32 -> bf16 P.  One block per row.
// ---------------------------------------------------------------------------
__global__ __launch_bounds__(256) void attn_softmax(
    const float* __restrict__ S, __bf16* __restrict__ P)
{
    __shared__ float red[256];
    const int row = blockIdx.x;
    const int tid = threadIdx.x;
    const float* __restrict__ s = S + (size_t)row * NTOK;

    float v[8];
    float mx = -3.4e38f;
#pragma unroll
    for (int i = 0; i < 8; ++i) {
        v[i] = s[tid + i * 256];
        mx = fmaxf(mx, v[i]);
    }
    red[tid] = mx;
    __syncthreads();
    for (int off = 128; off > 0; off >>= 1) {
        if (tid < off) red[tid] = fmaxf(red[tid], red[tid + off]);
        __syncthreads();
    }
    mx = red[0];
    __syncthreads();

    float sum = 0.f;
#pragma unroll
    for (int i = 0; i < 8; ++i) {
        v[i] = __expf(v[i] - mx);
        sum += v[i];
    }
    red[tid] = sum;
    __syncthreads();
    for (int off = 128; off > 0; off >>= 1) {
        if (tid < off) red[tid] += red[tid + off];
        __syncthreads();
    }
    const float inv = 1.0f / red[0];

    __bf16* __restrict__ p = P + (size_t)row * NTOK;
#pragma unroll
    for (int i = 0; i < 8; ++i) p[tid + i * 256] = f2bf(v[i] * inv);
}

// ---------------------------------------------------------------------------
// Kernel 4: out = P @ V, f32 out.  B-operand from transposed Vt[b][d][tok]:
// fixed d per lane, k (=token) contiguous.  32x64 tile per wave, K = 2048.
// ---------------------------------------------------------------------------
__global__ __launch_bounds__(256) void attn_pv(
    const __bf16* __restrict__ P, const __bf16* __restrict__ Vt,
    float* __restrict__ out)
{
    const int lane = threadIdx.x & 31;
    const int wave = threadIdx.x >> 5;

    const int TPB   = (NTOK / 32) * (DOUT / 64);    // 1024 tiles per batch
    const int t     = blockIdx.x * 8 + wave;
    const int batch = t / TPB;
    const int rem   = t % TPB;
    const int tm    = rem >> 4;                     // 0..63
    const int tn    = rem & 15;                     // 0..15

    const int m15   = lane & 15;
    const int khalf = (lane >> 4) << 3;

    const __bf16* __restrict__ arow0 =
        P + (size_t)(batch * NTOK + tm * 32      + m15) * NTOK + khalf;
    const __bf16* __restrict__ arow1 =
        P + (size_t)(batch * NTOK + tm * 32 + 16 + m15) * NTOK + khalf;

    v8f acc[2][4] = {};
    for (int k = 0; k < NTOK; k += 32) {
        __builtin_prefetch(arow0 + k + 128, 0, 0);
        v16bf a0 = load_frag_bf16(arow0 + k);
        v16bf a1 = load_frag_bf16(arow1 + k);
#pragma unroll
        for (int j = 0; j < 4; ++j) {
            const int d = tn * 64 + j * 16 + m15;
            const __bf16* __restrict__ brow =
                Vt + ((size_t)batch * DOUT + d) * NTOK + khalf + k;
            v16bf b = load_frag_bf16(brow);
            acc[0][j] = wmma_bf16(a0, b, acc[0][j]);
            acc[1][j] = wmma_bf16(a1, b, acc[1][j]);
        }
    }

#pragma unroll
    for (int i = 0; i < 2; ++i) {
#pragma unroll
        for (int j = 0; j < 4; ++j) {
            const int d = tn * 64 + j * 16 + m15;
#pragma unroll
            for (int r = 0; r < 8; ++r) {
                const int ir = tm * 32 + i * 16 + khalf + r;
                out[((size_t)batch * NTOK + ir) * DOUT + d] = acc[i][j][r];
            }
        }
    }
}

// ---------------------------------------------------------------------------
// Workspace layout (112 MB total):
//   [  0 .. 16M) Qb  bf16 [8192][1024]        (reused as P low half later)
//   [ 16 .. 32M) Kb  bf16 [8192][1024]        (reused as P high half later)
//   [ 32 .. 48M) Vt  bf16 [4][1024][2048]
//   [ 48 ..112M) S   f32  [4][2048][2048]
//     staging (dead before S is written):
//     [48 .. 64M) xb  bf16 [8192][1024]
//     [64 .. 66M) Wqb, [66 .. 68M) Wkb, [68 .. 70M) Wvb  (bf16 [1024][1024])
//   P bf16 [4][2048][2048] overwrites [0..32M) after scores are consumed.
// ---------------------------------------------------------------------------
extern "C" void kernel_launch(void* const* d_in, const int* in_sizes, int n_in,
                              void* d_out, int out_size, void* d_ws, size_t ws_size,
                              hipStream_t stream) {
    (void)in_sizes; (void)n_in; (void)out_size; (void)ws_size;
    const float* x  = (const float*)d_in[0];
    const float* Wk = (const float*)d_in[1];
    const float* Wq = (const float*)d_in[2];
    const float* Wv = (const float*)d_in[3];
    float* out = (float*)d_out;

    char* ws = (char*)d_ws;
    const size_t MB = 1024u * 1024u;
    __bf16* Qb  = (__bf16*)(ws + 0 * MB);
    __bf16* Kb  = (__bf16*)(ws + 16 * MB);
    __bf16* Vt  = (__bf16*)(ws + 32 * MB);
    float*  S   = (float*) (ws + 48 * MB);
    __bf16* xb  = (__bf16*)(ws + 48 * MB);
    __bf16* Wqb = (__bf16*)(ws + 64 * MB);
    __bf16* Wkb = (__bf16*)(ws + 66 * MB);
    __bf16* Wvb = (__bf16*)(ws + 68 * MB);
    __bf16* P   = (__bf16*)(ws + 0 * MB);   // reuse Qb/Kb region

    // 0) stage inputs as bf16 (one-shot conversion, native v_cvt path)
    cvt_f32_to_bf16<<<(MTOT * DIN / 8) / 256, 256, 0, stream>>>(x, xb, MTOT * DIN / 8);
    cvt_f32_to_bf16<<<(DOUT * DIN / 8) / 256, 256, 0, stream>>>(Wq, Wqb, DOUT * DIN / 8);
    cvt_f32_to_bf16<<<(DOUT * DIN / 8) / 256, 256, 0, stream>>>(Wk, Wkb, DOUT * DIN / 8);
    cvt_f32_to_bf16<<<(DOUT * DIN / 8) / 256, 256, 0, stream>>>(Wv, Wvb, DOUT * DIN / 8);

    // 1) QKV projections: 4096 wave-tiles each, 8 waves/block, y = which
    attn_qkv_proj<<<dim3(512, 3), 256, 0, stream>>>(xb, Wqb, Wkb, Wvb, Qb, Kb, Vt);
    // 2) scores: 4 batches * 2048 tiles = 8192 waves -> 1024 blocks
    attn_scores<<<1024, 256, 0, stream>>>(Qb, Kb, S);
    // 3) softmax: one block per row (B*N = 8192)
    attn_softmax<<<8192, 256, 0, stream>>>(S, P);
    // 4) P @ V: 4 * 1024 tiles = 4096 waves -> 512 blocks
    attn_pv<<<512, 256, 0, stream>>>(P, Vt, out);
}
